// MultiHeadAttention_65738769433381
// MI455X (gfx1250) — matrix operand order repaired and checked by reading, not compiled, not run on hardware
//
#include <hip/hip_runtime.h>
#include <hip/hip_bf16.h>
#include <math.h>
#include <stdint.h>

// Problem constants (match reference)
#define Bc  4
#define Sc  2048
#define Dc  512
#define Hc  8
#define DKc 64

// Fused-attention tiling
#define QT  16                       // queries per workgroup
#define RS  2052                     // padded LDS row stride in floats (2052 % 64 == 4)
#define SMEM_FLOATS (QT * RS + 4 * 256)
#define SMEM_BYTES  (SMEM_FLOATS * 4)   // 135,424 B  (2 blocks / 320KB WGP)

// Projection-kernel W-tile staging: 64 rows x 32 f32, rows padded to 36 f32
// (36*4 = 144 B -> every row 16B-aligned; 36 mod 64 rotation spreads banks)
#define WROW 36

typedef __attribute__((ext_vector_type(16))) __bf16 v16bf;
typedef __attribute__((ext_vector_type(8)))  float  v8f;

__device__ __forceinline__ v8f wmma_bf16(v16bf a, v16bf b, v8f c) {
  return __builtin_amdgcn_wmma_f32_16x16x32_bf16(false, a, false, b,
                                                 (short)0, c, false, false);
}

// ---------------------------------------------------------------------------
// gfx1250 async global->LDS copy (ISA 10.x GLOBAL_LOAD_ASYNC_TO_LDS_B128,
// tracked by ASYNCcnt). vdst = VGPR holding LDS byte address; GV addressing.
// ---------------------------------------------------------------------------
__device__ __forceinline__ void async_copy16(uint32_t lds_addr, const float* g) {
  asm volatile("global_load_async_to_lds_b128 %0, %1, off"
               :: "v"(lds_addr), "v"((uint64_t)(uintptr_t)g)
               : "memory");
}
#define WAIT_ASYNC_LE2() asm volatile("s_wait_asynccnt 0x2" ::: "memory")
#define WAIT_ASYNC_0()   asm volatile("s_wait_asynccnt 0x0" ::: "memory")

// ---------------------------------------------------------------------------
// WMMA fragment loaders (contiguous k order used by both A and B, wave32)
// A 16x32 bf16: lanes 0-15: M=lane, K 0..15;
//               lanes 16-31: K 16..31.
// ---------------------------------------------------------------------------
__device__ __forceinline__ v16bf load_a_f32(const float* __restrict__ src, int ld,
                                            int m0, int k0, int lane) {
  const int half = lane >> 4;
  const int m    = lane & 15;
  const float* p = src + (size_t)(m0 + m) * ld + k0 + half * 16;
  v16bf a;
#pragma unroll
  for (int i = 0; i < 8; ++i) a[i]     = (__bf16)p[i];
#pragma unroll
  for (int i = 0; i < 8; ++i) a[8 + i] = (__bf16)p[8 + i];
  return a;
}

__device__ __forceinline__ v16bf load_a_bf16(const __bf16* __restrict__ src, int ld,
                                             int m0, int k0, int lane) {
  const int half = lane >> 4;
  const int m    = lane & 15;
  const __bf16* p = src + (size_t)(m0 + m) * ld + k0 + half * 16;
  v16bf a;
#pragma unroll
  for (int i = 0; i < 8; ++i) a[i]     = p[i];
#pragma unroll
  for (int i = 0; i < 8; ++i) a[8 + i] = p[8 + i];
  return a;
}

// B 32x16 bf16 where column n of B is row n of a row-major source.
// lanes 0-15: col n, K 0..15 in v0-7; lanes 16-31: col n, K 16..31.
__device__ __forceinline__ v16bf load_bT_bf16(const __bf16* __restrict__ src, int ld,
                                              int n0, int k0, int lane) {
  const int half = lane >> 4;
  const int n    = lane & 15;
  const __bf16* p = src + (size_t)(n0 + n) * ld + k0 + half * 16;
  v16bf b;
#pragma unroll
  for (int i = 0; i < 16; ++i) b[i] = p[i];
  return b;
}

// B fragment read from LDS-staged f32 W tile (rows padded to WROW floats).
__device__ __forceinline__ v16bf load_bT_lds(const float* buf, int nloc, int lane) {
  const int half = lane >> 4;
  const int n    = nloc + (lane & 15);
  const float* p = buf + n * WROW + half * 16;
  v16bf b;
#pragma unroll
  for (int i = 0; i < 16; ++i) b[i] = (__bf16)p[i];
  return b;
}

// Stage a 64x32 f32 tile of W (rows n0b..n0b+63, cols k0..k0+31) into LDS.
// 512 b128 transfers, 2 per thread, all via ASYNCcnt.
__device__ __forceinline__ void stage_w_tile(float* buf, const float* __restrict__ W,
                                             int n0b, int k0, int tid) {
#pragma unroll
  for (int j = 0; j < 2; ++j) {
    const int i   = tid * 2 + j;        // 0..511
    const int row = i >> 3;             // 0..63
    const int c4  = (i & 7) * 4;        // 0..28 step 4
    const float* g = W + (size_t)(n0b + row) * Dc + k0 + c4;
    async_copy16((uint32_t)(uintptr_t)(buf + row * WROW + c4), g);
  }
}

// ---------------------------------------------------------------------------
// Kernel 1: QKV projection  y = x @ W.T + bias -> head layout (bf16).
// W tile double-buffered in LDS via async global->LDS copies.
// mode 0: dst[b,h,s,d] (Q,K)   mode 1: dst[b,h,d,s] (V transposed)
// ---------------------------------------------------------------------------
__global__ __launch_bounds__(256)
void qkv_proj_kernel(const float* __restrict__ X, const float* __restrict__ W,
                     const float* __restrict__ bias, __bf16* __restrict__ dst,
                     int mode) {
  __shared__ float Wt[2][64 * WROW];

  const int tid  = threadIdx.x;
  const int lane = tid & 31;
  const int wave = tid >> 5;
  const int m0  = blockIdx.x * 32 + (wave >> 2) * 16;
  const int n0b = blockIdx.y * 64;              // block's W-row range
  const int nloc = (wave & 3) * 16;             // wave's tile inside the stage

  stage_w_tile(Wt[0], W, n0b, 0, tid);

  v8f acc = {};
  for (int step = 0; step < Dc / 32; ++step) {  // 16 k-steps
    const int k0 = step * 32;
    if (step + 1 < Dc / 32) {
      stage_w_tile(Wt[(step + 1) & 1], W, n0b, k0 + 32, tid);
      WAIT_ASYNC_LE2();                         // step's 2 copies landed
    } else {
      WAIT_ASYNC_0();
    }
    __syncthreads();                            // block-wide visibility
    v16bf a = load_a_f32(X, Dc, m0, k0, lane);
    v16bf b = load_bT_lds(Wt[step & 1], nloc, lane);
    acc = wmma_bf16(a, b, acc);
    __syncthreads();                            // done reading before re-stage
  }

  const int half = lane >> 4;
  const int gn   = n0b + nloc + (lane & 15);
  const int h = gn >> 6, d = gn & 63;
  const float bval = bias[gn];
#pragma unroll
  for (int r = 0; r < 8; ++r) {
    const int gm = m0 + half * 8 + r;       // row in [0, B*S)
    const int bb = gm >> 11;                // / S
    const int s  = gm & (Sc - 1);
    const float v = acc[r] + bval;
    size_t idx;
    if (mode == 1) idx = (((size_t)bb * Hc + h) * DKc + d) * Sc + s;  // V^T
    else           idx = (((size_t)bb * Hc + h) * Sc + s) * DKc + d;  // Q,K
    dst[idx] = (__bf16)v;
  }
}

// ---------------------------------------------------------------------------
// Kernel 2 (fused attention): one 256-thread block owns 16 queries of one
// (b,h). Full 16x2048 score stripe lives in LDS (~132KB of the 320KB WGP LDS).
//   Phase 1: S = mask ? QK^T/8 : -1e9          (WMMA, global->LDS)
//   Phase 2: softmax rows in LDS; single write of final scores to d_out
//   Phase 3: context = P @ V                   (WMMA, A read from LDS)
// ---------------------------------------------------------------------------
__global__ __launch_bounds__(256)
void attn_fused_kernel(const __bf16* __restrict__ Qh, const __bf16* __restrict__ Kh,
                       const __bf16* __restrict__ Vt, const int* __restrict__ mask,
                       float* __restrict__ scores, __bf16* __restrict__ concat) {
  extern __shared__ float smem[];
  float* Pt   = smem;               // QT x RS padded score stripe
  float* comb = smem + QT * RS;     // 4 tiles x 256 floats (K-half combine)

  const int lane = threadIdx.x & 31;
  const int wave = threadIdx.x >> 5;
  const int bh = blockIdx.y;
  const int b = bh >> 3, h = bh & 7;
  const int q0 = blockIdx.x * QT;

  const __bf16* Q = Qh + (size_t)bh * Sc * DKc;
  const __bf16* K = Kh + (size_t)bh * Sc * DKc;
  const __bf16* V = Vt + (size_t)bh * DKc * Sc;
  const int* Mrow = mask + (size_t)b * Sc * Sc + (size_t)q0 * Sc;

  const int half = lane >> 4;
  const int col  = lane & 15;

  // ---- Phase 1: scores stripe into LDS ----------------------------------
  v16bf aQ0 = load_a_bf16(Q, DKc, q0, 0,  lane);
  v16bf aQ1 = load_a_bf16(Q, DKc, q0, 32, lane);

  for (int t = 0; t < 16; ++t) {
    const int n0 = wave * 256 + t * 16;        // this wave's key tile
    v8f acc = {};
    acc = wmma_bf16(aQ0, load_bT_bf16(K, DKc, n0, 0,  lane), acc);
    acc = wmma_bf16(aQ1, load_bT_bf16(K, DKc, n0, 32, lane), acc);
    const int c = n0 + col;
#pragma unroll
    for (int r = 0; r < 8; ++r) {
      const int m = half * 8 + r;
      const float sc = acc[r] * 0.125f;        // 1/sqrt(64)
      const int mv = Mrow[(size_t)m * Sc + c];
      Pt[m * RS + c] = (mv == 0) ? -1.0e9f : sc;
    }
  }
  __syncthreads();

  // ---- Phase 2: softmax per row (wave w owns rows 2w, 2w+1) -------------
  float* Srow = scores + ((size_t)bh * Sc + q0) * Sc;
#pragma unroll
  for (int rr = 0; rr < 2; ++rr) {
    const int r = wave * 2 + rr;
    float* row = Pt + r * RS;
    float m = -3.4e38f;
    for (int i = lane; i < Sc; i += 32) m = fmaxf(m, row[i]);
#pragma unroll
    for (int off = 16; off > 0; off >>= 1) m = fmaxf(m, __shfl_xor(m, off, 32));
    float s = 0.0f;
    for (int i = lane; i < Sc; i += 32) {
      const float e = __expf(row[i] - m);
      row[i] = e;
      s += e;
    }
#pragma unroll
    for (int off = 16; off > 0; off >>= 1) s += __shfl_xor(s, off, 32);
    const float inv = 1.0f / s;
    float* gout = Srow + (size_t)r * Sc;
    for (int i = lane; i < Sc; i += 32) {
      const float p = row[i] * inv;
      row[i] = p;                               // normalized P stays in LDS
      gout[i] = p;                              // single HBM write of scores
    }
  }
  __syncthreads();

  // ---- Phase 3: context = P @ V  (A from LDS, f32->bf16 in regs) --------
  const int ntile = wave & 3;                  // d-tile 0..3 (DK = 64)
  const int khalf = wave >> 2;                 // split K range across wave pairs
  const int n0 = ntile * 16;

  v8f acc = {};
  for (int k0 = khalf * 1024; k0 < khalf * 1024 + 1024; k0 += 32) {
    __builtin_prefetch(V + (size_t)(n0 + col) * Sc + k0 + 256, 0, 1);
    v16bf a;
    const float* pr = Pt + (size_t)col * RS + k0 + half * 16;  // A: M=col
#pragma unroll
    for (int i = 0; i < 8; ++i) a[i]     = (__bf16)pr[i];
#pragma unroll
    for (int i = 0; i < 8; ++i) a[8 + i] = (__bf16)pr[8 + i];
    v16bf bm = load_bT_bf16(V, Sc, n0, k0, lane);
    acc = wmma_bf16(a, bm, acc);
  }

  if (khalf == 1) {
#pragma unroll
    for (int r = 0; r < 8; ++r) comb[ntile * 256 + r * 32 + lane] = acc[r];
  }
  __syncthreads();
  if (khalf == 0) {
#pragma unroll
    for (int r = 0; r < 8; ++r) {
      const int m = half * 8 + r;
      const float v = acc[r] + comb[ntile * 256 + r * 32 + lane];
      concat[((size_t)b * Sc + q0 + m) * Dc + h * DKc + n0 + col] = (__bf16)v;
    }
  }
}

// ---------------------------------------------------------------------------
// Kernel 3: out = concat @ Wo.T + bo   (fp32 out region of d_out)
// Same async double-buffered W staging as the projection kernel.
// ---------------------------------------------------------------------------
__global__ __launch_bounds__(256)
void out_proj_kernel(const __bf16* __restrict__ Xc, const float* __restrict__ Wo,
                     const float* __restrict__ bo, float* __restrict__ out) {
  __shared__ float Wt[2][64 * WROW];

  const int tid  = threadIdx.x;
  const int lane = tid & 31;
  const int wave = tid >> 5;
  const int m0  = blockIdx.x * 32 + (wave >> 2) * 16;
  const int n0b = blockIdx.y * 64;
  const int nloc = (wave & 3) * 16;

  stage_w_tile(Wt[0], Wo, n0b, 0, tid);

  v8f acc = {};
  for (int step = 0; step < Dc / 32; ++step) {
    const int k0 = step * 32;
    if (step + 1 < Dc / 32) {
      stage_w_tile(Wt[(step + 1) & 1], Wo, n0b, k0 + 32, tid);
      WAIT_ASYNC_LE2();
    } else {
      WAIT_ASYNC_0();
    }
    __syncthreads();
    v16bf a = load_a_bf16(Xc, Dc, m0, k0, lane);
    v16bf b = load_bT_lds(Wt[step & 1], nloc, lane);
    acc = wmma_bf16(a, b, acc);
    __syncthreads();
  }

  const int half = lane >> 4;
  const int gn = n0b + nloc + (lane & 15);
  const float bval = bo[gn];
#pragma unroll
  for (int r = 0; r < 8; ++r) {
    const int gm = m0 + half * 8 + r;
    out[(size_t)gm * Dc + gn] = acc[r] + bval;
  }
}

// ---------------------------------------------------------------------------
extern "C" void kernel_launch(void* const* d_in, const int* in_sizes, int n_in,
                              void* d_out, int out_size, void* d_ws, size_t ws_size,
                              hipStream_t stream) {
  const float* q    = (const float*)d_in[0];
  const float* k    = (const float*)d_in[1];
  const float* v    = (const float*)d_in[2];
  const int*   mask = (const int*)  d_in[3];
  const float* Wq   = (const float*)d_in[4];
  const float* bq   = (const float*)d_in[5];
  const float* Wk   = (const float*)d_in[6];
  const float* bk   = (const float*)d_in[7];
  const float* Wv   = (const float*)d_in[8];
  const float* bv   = (const float*)d_in[9];
  const float* Wo   = (const float*)d_in[10];
  const float* bo   = (const float*)d_in[11];

  float* out    = (float*)d_out;                   // [B,S,D]
  float* scores = out + (size_t)Bc * Sc * Dc;      // [B,H,S,S]

  const size_t NE = (size_t)Bc * Hc * Sc * DKc;    // == B*S*D
  __bf16* Qh = (__bf16*)d_ws;      // [B,H,S,DK]
  __bf16* Kh = Qh + NE;            // [B,H,S,DK]
  __bf16* Vt = Kh + NE;            // [B,H,DK,S]
  __bf16* Xc = Vt + NE;            // concat [B,S,D]

  // Allow >64KB dynamic LDS for the fused kernel (immediate host-side call;
  // not a stream op, safe under graph capture, deterministic).
  (void)hipFuncSetAttribute((const void*)attn_fused_kernel,
                            hipFuncAttributeMaxDynamicSharedMemorySize,
                            SMEM_BYTES);

  dim3 blk(256);

  dim3 gproj(Bc * Sc / 32, Dc / 64);               // (256, 8)
  qkv_proj_kernel<<<gproj, blk, 0, stream>>>(q, Wq, bq, Qh, 0);
  qkv_proj_kernel<<<gproj, blk, 0, stream>>>(k, Wk, bk, Kh, 0);
  qkv_proj_kernel<<<gproj, blk, 0, stream>>>(v, Wv, bv, Vt, 1);

  dim3 gattn(Sc / QT, Bc * Hc);                    // (128, 32)
  attn_fused_kernel<<<gattn, blk, SMEM_BYTES, stream>>>(Qh, Kh, Vt, mask,
                                                        scores, Xc);

  dim3 gout(Bc * Sc / 32, Dc / 64);                // (256, 8)
  out_proj_kernel<<<gout, blk, 0, stream>>>(Xc, Wo, bo, out);
}
